// ImplicitLayer_61873298866710
// MI455X (gfx1250) — matrix operand order repaired and checked by reading, not compiled, not run on hardware
//
#include <hip/hip_runtime.h>

// ---------------------------------------------------------------------------
// Implicit (DEQ) layer, fused for MI455X / gfx1250 (wave32, WMMA).
//
//   ux  = x @ W_in^T                         [4096,1024]
//   z   = 0 ; 50x: z = tanh(z @ W_imp^T + ux)
//   out = z @ W_out^T                        [4096,10]
//
// Row independence of the fixed point => each workgroup owns MT=32 batch
// rows, keeps ux resident in VGPRs, z in double-buffered LDS (bf16, 128 KB),
// and streams L2-resident bf16 weights through v_wmma_f32_16x16x32_bf16.
// One __syncthreads per iteration; no grid-wide sync needed.
// ---------------------------------------------------------------------------

typedef __bf16 bf16_t;
typedef __attribute__((ext_vector_type(4)))  bf16_t v4bf;
typedef __attribute__((ext_vector_type(8)))  bf16_t v8bf;
typedef __attribute__((ext_vector_type(16))) bf16_t v16bf;
typedef __attribute__((ext_vector_type(8)))  float  v8f;
typedef __attribute__((ext_vector_type(4)))  float  v4f;

#define BATCH   4096
#define INFEAT  3072
#define WIDTH   1024
#define OUTF    10
#define ITERS   50

#define MT      32                 // batch rows per workgroup
#define NWAVES  16                 // waves per workgroup (512 threads)
#define NPW     (WIDTH / NWAVES)   // 64 output columns per wave
#define MTILES  (MT / 16)          // 2
#define NTILES  (NPW / 16)         // 4

// tanh(v) = 1 - 2/(exp2(2*log2(e)*v) + 1); saturates correctly at +/-1.
__device__ __forceinline__ float fast_tanh(float v) {
    float e = __builtin_amdgcn_exp2f(v * 2.885390081777927f); // e^(2v)
    return 1.0f - 2.0f * __builtin_amdgcn_rcpf(e + 1.0f);
}

__device__ __forceinline__ v16bf cat8(v8bf lo, v8bf hi) {
    return __builtin_shufflevector(lo, hi, 0, 1, 2, 3, 4, 5, 6, 7,
                                           8, 9, 10, 11, 12, 13, 14, 15);
}

// f32 -> bf16, vectorized (float4 in, 4x bf16 out)
extern "C" __global__ void cvt_bf16_kernel(const float* __restrict__ s,
                                           bf16_t* __restrict__ d, int n4) {
    int i = blockIdx.x * blockDim.x + threadIdx.x;
    int stride = gridDim.x * blockDim.x;
    const v4f* sv = (const v4f*)s;
    v4bf* dv = (v4bf*)d;
    for (; i < n4; i += stride) {
        v4f f = sv[i];
        v4bf b;
#pragma unroll
        for (int j = 0; j < 4; ++j) b[j] = (bf16_t)f[j];
        dv[i] = b;
    }
}

template <bool XBF>  // true: x pre-converted to bf16 in workspace
__global__ __launch_bounds__(NWAVES * 32, 1)
void deq_fused_kernel(const float*  __restrict__ x,     // [4096,3072] f32
                      const bf16_t* __restrict__ xb,    // [4096,3072] bf16 (if XBF)
                      const bf16_t* __restrict__ Winb,  // [1024,3072] bf16
                      const bf16_t* __restrict__ Wimb,  // [1024,1024] bf16
                      const float*  __restrict__ Wout,  // [10,1024]   f32
                      float*        __restrict__ out)   // [4096,10]   f32
{
    extern __shared__ char smem_raw[];
    // double-buffered z tile: [2][MT][WIDTH] bf16 = 128 KB
    bf16_t (*zs)[MT][WIDTH] = reinterpret_cast<bf16_t (*)[MT][WIDTH]>(smem_raw);

    const int m0   = blockIdx.x * MT;
    const int lane = threadIdx.x & 31;
    const int wave = threadIdx.x >> 5;
    const int lhi  = lane >> 4;     // lane half selects K sub-range
    const int llo  = lane & 15;     // row (A) / column (B,C) within tile
    const int n0   = wave * NPW;    // this wave's output-column slice
    const int ak   = 8 * lhi;       // A: lanes 0-15 -> K 0-7 & 16-23; 16-31 -> 8-15 & 24-31
    const int bk   = 8 * lhi;       // B: lanes 0-15 -> K 0-7 & 16-23; 16-31 -> 8-15 & 24-31

    v8f acc[MTILES][NTILES] = {};   // WMMA C/D fragments
    v8f uxr[MTILES][NTILES];        // ux kept resident in VGPRs

    // ---------------- GEMM1: ux = x @ W_in^T  (K = 3072) ----------------
    const float*  xrow[MTILES];
    const bf16_t* xbrow[MTILES];
    const bf16_t* wrow[NTILES];
    const bf16_t* imrow[NTILES];
#pragma unroll
    for (int mt = 0; mt < MTILES; ++mt) {
        const size_t r = (size_t)(m0 + 16 * mt + llo) * INFEAT;
        xrow[mt]  = x + r;
        xbrow[mt] = XBF ? (xb + r) : nullptr;
    }
#pragma unroll
    for (int nt = 0; nt < NTILES; ++nt) {
        wrow[nt]  = Winb + (size_t)(n0 + 16 * nt + llo) * INFEAT;
        imrow[nt] = Wimb + (size_t)(n0 + 16 * nt + llo) * WIDTH;
    }

    for (int kc = 0; kc < INFEAT; kc += 32) {
        v16bf A[MTILES];
#pragma unroll
        for (int mt = 0; mt < MTILES; ++mt) {
            if constexpr (XBF) {
                const v8bf* p0 = (const v8bf*)(xbrow[mt] + kc + ak);
                const v8bf* p1 = (const v8bf*)(xbrow[mt] + kc + ak + 16);
                A[mt] = cat8(p0[0], p1[0]);
            } else {
                // two contiguous 32B f32 chunks per lane, converted to bf16
                const v4f* p0 = (const v4f*)(xrow[mt] + kc + ak);
                const v4f* p1 = (const v4f*)(xrow[mt] + kc + ak + 16);
                v4f f0 = p0[0], f1 = p0[1], f2 = p1[0], f3 = p1[1];
                v16bf a;
#pragma unroll
                for (int i = 0; i < 4; ++i) {
                    a[i]      = (bf16_t)f0[i];
                    a[i + 4]  = (bf16_t)f1[i];
                    a[i + 8]  = (bf16_t)f2[i];
                    a[i + 12] = (bf16_t)f3[i];
                }
                A[mt] = a;
            }
        }
#pragma unroll
        for (int nt = 0; nt < NTILES; ++nt) {
            const v8bf* bp = (const v8bf*)(wrow[nt] + kc + bk);
            v16bf Bt = cat8(bp[0], bp[2]);
#pragma unroll
            for (int mt = 0; mt < MTILES; ++mt)
                acc[mt][nt] = __builtin_amdgcn_wmma_f32_16x16x32_bf16(
                    false, A[mt], false, Bt, (short)0, acc[mt][nt], false, false);
        }
    }

    // z1 = tanh(ux); stash ux in registers, z1 (bf16) into LDS buffer 0
#pragma unroll
    for (int mt = 0; mt < MTILES; ++mt)
#pragma unroll
        for (int nt = 0; nt < NTILES; ++nt) {
            uxr[mt][nt] = acc[mt][nt];
#pragma unroll
            for (int r = 0; r < 8; ++r) {
                float zv = fast_tanh(acc[mt][nt][r]);
                zs[0][16 * mt + 8 * lhi + r][n0 + 16 * nt + llo] = (bf16_t)zv;
            }
        }
    __syncthreads();

    // ------------- 49 recurrent iterations: z = tanh(z @ W_imp^T + ux) -------------
    for (int it = 1; it < ITERS; ++it) {
        const int rb = (it - 1) & 1;
        const int wb = it & 1;

        v8f a2[MTILES][NTILES] = {};
        for (int kc = 0; kc < WIDTH; kc += 32) {
            v16bf A[MTILES];
#pragma unroll
            for (int mt = 0; mt < MTILES; ++mt) {
                const v8bf* ap0 = (const v8bf*)&zs[rb][16 * mt + llo][kc + ak];
                const v8bf* ap1 = (const v8bf*)&zs[rb][16 * mt + llo][kc + ak + 16];
                A[mt] = cat8(ap0[0], ap1[0]);
            }
#pragma unroll
            for (int nt = 0; nt < NTILES; ++nt) {
                const v8bf* bp = (const v8bf*)(imrow[nt] + kc + bk);
                v16bf Bt = cat8(bp[0], bp[2]);
#pragma unroll
                for (int mt = 0; mt < MTILES; ++mt)
                    a2[mt][nt] = __builtin_amdgcn_wmma_f32_16x16x32_bf16(
                        false, A[mt], false, Bt, (short)0, a2[mt][nt], false, false);
            }
        }

#pragma unroll
        for (int mt = 0; mt < MTILES; ++mt)
#pragma unroll
            for (int nt = 0; nt < NTILES; ++nt)
#pragma unroll
                for (int r = 0; r < 8; ++r) {
                    float zv = fast_tanh(a2[mt][nt][r] + uxr[mt][nt][r]);
                    zs[wb][16 * mt + 8 * lhi + r][n0 + 16 * nt + llo] = (bf16_t)zv;
                }
        __syncthreads();
    }

    // ------------- readout: out = z @ W_out^T (32x10 per WG, vectorized) -------------
    const int last = (ITERS - 1) & 1;
    const int t = threadIdx.x;
    if (t < MT * OUTF) {
        const int m = t / OUTF;
        const int o = t % OUTF;
        const v8bf* zrow = (const v8bf*)&zs[last][m][0];
        const v4f*  wo   = (const v4f*)(Wout + (size_t)o * WIDTH);
        float s = 0.0f;
        for (int c = 0; c < WIDTH / 8; ++c) {
            v8bf zv = zrow[c];
            v4f w0 = wo[2 * c], w1 = wo[2 * c + 1];
#pragma unroll
            for (int i = 0; i < 4; ++i) {
                s += (float)zv[i]     * w0[i];
                s += (float)zv[i + 4] * w1[i];
            }
        }
        out[(size_t)(m0 + m) * OUTF + o] = s;
    }
}

extern "C" void kernel_launch(void* const* d_in, const int* in_sizes, int n_in,
                              void* d_out, int out_size, void* d_ws, size_t ws_size,
                              hipStream_t stream) {
    (void)in_sizes; (void)n_in; (void)out_size;
    const float* x     = (const float*)d_in[0];  // [4096,3072]
    const float* W_in  = (const float*)d_in[1];  // [1024,3072]
    const float* W_imp = (const float*)d_in[2];  // [1024,1024]
    const float* W_out = (const float*)d_in[3];  // [10,1024]
    float* out = (float*)d_out;

    char* ws = (char*)d_ws;
    const size_t nWin = (size_t)WIDTH * INFEAT;          // 3.1M
    const size_t nWim = (size_t)WIDTH * WIDTH;           // 1.0M
    const size_t nX   = (size_t)BATCH * INFEAT;          // 12.6M
    bf16_t* Winb = (bf16_t*)ws;                          // 6 MB
    bf16_t* Wimb = (bf16_t*)(ws + nWin * 2);             // 2 MB
    bf16_t* xbw  = (bf16_t*)(ws + (nWin + nWim) * 2);    // 25 MB (optional)
    const bool use_xb = ws_size >= (nWin + nWim + nX) * 2;

    cvt_bf16_kernel<<<(int)((nWin / 4 + 255) / 256), 256, 0, stream>>>(W_in, Winb, (int)(nWin / 4));
    cvt_bf16_kernel<<<(int)((nWim / 4 + 255) / 256), 256, 0, stream>>>(W_imp, Wimb, (int)(nWim / 4));

    const size_t smem = (size_t)2 * MT * WIDTH * sizeof(bf16_t); // 128 KB (<= 320 KB WGP LDS)
    if (use_xb) {
        cvt_bf16_kernel<<<(int)((nX / 4 + 255) / 256), 256, 0, stream>>>(x, xbw, (int)(nX / 4));
        deq_fused_kernel<true><<<BATCH / MT, NWAVES * 32, smem, stream>>>(
            x, xbw, Winb, Wimb, W_out, out);
    } else {
        deq_fused_kernel<false><<<BATCH / MT, NWAVES * 32, smem, stream>>>(
            x, nullptr, Winb, Wimb, W_out, out);
    }
}
